// LinearAttention_29850022707449
// MI455X (gfx1250) — matrix operand order repaired and checked by reading, not compiled, not run on hardware
//
#include <hip/hip_runtime.h>

// LinearAttention for MI455X (gfx1250): all GEMMs via v_wmma_f32_16x16x32_bf16
//   x:(8,512,64,64) f32, w_qkv:(1536,512), w_out:(512,512), b_out:(512)
//   out:(8,512,64,64) f32

#define NPIX 4096        // 64*64
#define CDIM 512
#define QKVC 1536

typedef __attribute__((ext_vector_type(16))) __bf16 v16bf;
typedef __attribute__((ext_vector_type(8)))  __bf16 v8bf;
typedef __attribute__((ext_vector_type(8)))  float  v8f;

union BF16x8 { v8bf v; __bf16 s[8]; };
union Frag16 { v16bf v; v8bf h[2]; };

__device__ __forceinline__ v16bf frag_from(const __bf16* lo, const __bf16* hi) {
  Frag16 f;
  f.h[0] = *(const v8bf*)lo;
  f.h[1] = *(const v8bf*)hi;
  return f.v;
}

__device__ __forceinline__ v8f wmma_bf16(v16bf a, v16bf b, v8f c) {
  // (neg_a, A, neg_b, B, c_mod, C, reuse_a, reuse_b)
  return __builtin_amdgcn_wmma_f32_16x16x32_bf16(false, a, false, b, (short)0, c,
                                                 false, false);
}

__device__ __forceinline__ v8f zero8() {
  v8f z;
  for (int j = 0; j < 8; ++j) z[j] = 0.0f;
  return z;
}

// ---------------------------------------------------------------------------
// Generic GEMM: C[m,n] = sum_k A[m,k]*B[k,n]
// A: f32 row-major MxK (batch stride strideA, elements).
// B: row-major KxN (batch stride strideB, elements); f32 or bf16 (B_BF16).
// C: bf16, or f32 + bias (F32_OUT).
// 128x128 tile per workgroup, 8 waves, K staged 32-wide through LDS in bf16.
// ---------------------------------------------------------------------------
template <bool F32_OUT, bool B_BF16>
__global__ __launch_bounds__(256) void wmma_gemm(
    const float* __restrict__ A, long long strideA,
    const void* __restrict__ B, long long strideB,
    void* __restrict__ Cv, long long strideC,
    const float* __restrict__ bias, int K, int N) {
  __shared__ __bf16 sA[128 * 40];  // [m][k], padded stride 40 (80B, 16B-mult)
  __shared__ __bf16 sB[128 * 40];  // [n][k] (transposed), padded stride 40

  const int tid = threadIdx.x;
  const int lane = tid & 31, wave = tid >> 5;
  const int half = lane >> 4, l15 = lane & 15;
  const int bz = blockIdx.z;
  const int m0 = blockIdx.y * 128, n0 = blockIdx.x * 128;

  const float*  Ab  = A + (long long)bz * strideA;
  const float*  BbF = (const float*)B + (long long)bz * strideB;
  const __bf16* BbH = (const __bf16*)B + (long long)bz * strideB;

  const int wm = (wave & 3) * 32;   // wave's 32-row block
  const int wn = (wave >> 2) * 64;  // wave's 64-col block
  v8f acc[2][4];
  for (int i = 0; i < 2; ++i)
    for (int j = 0; j < 4; ++j) acc[i][j] = zero8();

  for (int k0 = 0; k0 < K; k0 += 32) {
    // stage A tile 128x32 (f32 -> bf16)
    for (int i = 0; i < 4; ++i) {
      int r = (tid >> 3) + i * 32;
      int kk = (tid & 7) * 4;
      float4 f = *(const float4*)&Ab[(long long)(m0 + r) * K + (k0 + kk)];
      __bf16* d = &sA[r * 40 + kk];
      d[0] = (__bf16)f.x; d[1] = (__bf16)f.y;
      d[2] = (__bf16)f.z; d[3] = (__bf16)f.w;
    }
    // stage B tile 32x128 transposed into sB[n][k]
    if constexpr (B_BF16) {
      for (int i = 0; i < 2; ++i) {
        int kk = (tid >> 4) + i * 16;       // 0..31
        int nn = (tid & 15) * 8;            // 0..120
        const __bf16* src = &BbH[(long long)(k0 + kk) * N + (n0 + nn)];
        if (k0 + 32 < K)
          __builtin_prefetch(src + 32LL * N, 0, 1);
        BF16x8 u; u.v = *(const v8bf*)src;
        for (int j = 0; j < 8; ++j) sB[(nn + j) * 40 + kk] = u.s[j];
      }
    } else {
      for (int i = 0; i < 4; ++i) {
        int kk = (tid >> 5) + i * 8;
        int nn = (tid & 31) * 4;
        const float* src = &BbF[(long long)(k0 + kk) * N + (n0 + nn)];
        if (k0 + 32 < K)  // prefetch next K-stage of B (global_prefetch_b8)
          __builtin_prefetch(src + 32LL * N, 0, 1);
        float4 f = *(const float4*)src;
        sB[(nn + 0) * 40 + kk] = (__bf16)f.x;
        sB[(nn + 1) * 40 + kk] = (__bf16)f.y;
        sB[(nn + 2) * 40 + kk] = (__bf16)f.z;
        sB[(nn + 3) * 40 + kk] = (__bf16)f.w;
      }
    }
    __syncthreads();

    v16bf af[2], bfr[4];
    for (int mi = 0; mi < 2; ++mi) {
      const __bf16* base = &sA[(wm + mi * 16 + l15) * 40];
      // A lane layout: lanes0-15 K={0..7,16..23}, lanes16-31 K={8..15,24..31}
      af[mi] = frag_from(base + half * 8, base + 16 + half * 8);
    }
    for (int ni = 0; ni < 4; ++ni) {
      const __bf16* base = &sB[(wn + ni * 16 + l15) * 40];
      // B lane layout: lanes0-15 K={0..7,16..23}, lanes16-31 K={8..15,24..31}, column = l15
      bfr[ni] = frag_from(base + half * 8, base + 16 + half * 8);
    }
    for (int mi = 0; mi < 2; ++mi)
      for (int ni = 0; ni < 4; ++ni)
        acc[mi][ni] = wmma_bf16(af[mi], bfr[ni], acc[mi][ni]);
    __syncthreads();
  }

  // epilogue: D vgpr j -> row half*8+j, col l15
  for (int mi = 0; mi < 2; ++mi)
    for (int ni = 0; ni < 4; ++ni) {
      int r = m0 + wm + mi * 16 + half * 8;
      int c = n0 + wn + ni * 16 + l15;
      for (int j = 0; j < 8; ++j) {
        long long idx = (long long)(r + j) * N + c + (long long)bz * strideC;
        float v = acc[mi][ni][j];
        if constexpr (F32_OUT)
          ((float*)Cv)[idx] = v + bias[r + j];
        else
          ((__bf16*)Cv)[idx] = (__bf16)v;
      }
    }
}

// ---------------------------------------------------------------------------
// In-place softmax over n (4096) for the K part of qkv. One wave per row.
// qkv layout: [b][1536][4096] bf16; k rows are channels 512..1023.
// ---------------------------------------------------------------------------
__global__ __launch_bounds__(256) void softmax_inplace(__bf16* __restrict__ qkv) {
  const int lane = threadIdx.x & 31, wave = threadIdx.x >> 5;
  const int row = blockIdx.x * 8 + wave;  // 0..4095 = 8 batches * 512 chans
  const int b = row >> 9, c = row & 511;
  __bf16* p = qkv + ((long long)b * QKVC + 512 + c) * NPIX;

  float m = -3.0e38f;
  for (int i = lane * 8; i < NPIX; i += 32 * 8) {
    BF16x8 u; u.v = *(const v8bf*)&p[i];
    for (int j = 0; j < 8; ++j) m = fmaxf(m, (float)u.s[j]);
  }
  for (int s = 16; s > 0; s >>= 1) m = fmaxf(m, __shfl_xor(m, s, 32));

  float sum = 0.0f;
  for (int i = lane * 8; i < NPIX; i += 32 * 8) {
    BF16x8 u; u.v = *(const v8bf*)&p[i];
    for (int j = 0; j < 8; ++j) sum += __expf((float)u.s[j] - m);
  }
  for (int s = 16; s > 0; s >>= 1) sum += __shfl_xor(sum, s, 32);
  const float inv = 1.0f / sum;

  for (int i = lane * 8; i < NPIX; i += 32 * 8) {
    BF16x8 u; u.v = *(const v8bf*)&p[i];
    for (int j = 0; j < 8; ++j) u.s[j] = (__bf16)(__expf((float)u.s[j] - m) * inv);
    *(v8bf*)&p[i] = u.v;
  }
}

// ---------------------------------------------------------------------------
// Per-(b,h): context[d,e] = sum_n k[d,n]*v[e,n]  (64x64, K=4096)
// then      out2[e,n]    = sum_d context[d,e]*q[d,n]  (64x4096, K=64)
// k/v rows are K-contiguous -> WMMA fragments load straight from global.
// ---------------------------------------------------------------------------
__global__ __launch_bounds__(256) void attn_fused(const __bf16* __restrict__ qkv,
                                                  __bf16* __restrict__ out2) {
  __shared__ __bf16 sCtx[64 * 72];   // context^T [e][d], stride 72 (144B)
  __shared__ __bf16 sQ[128 * 72];    // q tile transposed [n][d]

  const int tid = threadIdx.x, lane = tid & 31, wave = tid >> 5;
  const int half = lane >> 4, l15 = lane & 15;
  const int bh = blockIdx.x, b = bh >> 3, h = bh & 7;

  const __bf16* qp = qkv + ((long long)b * QKVC + h * 64) * NPIX;
  const __bf16* kp = qp + (long long)512 * NPIX;
  const __bf16* vp = qp + (long long)1024 * NPIX;

  // ---- phase 1: context = k_sm @ v^T  (each wave: one d-block, two e-blocks)
  const int dblk = wave >> 1;
  const int eb0 = (wave & 1) * 2;
  v8f acc[2];
  acc[0] = zero8(); acc[1] = zero8();

  const __bf16* arow  = kp + (long long)(dblk * 16 + l15) * NPIX;
  const __bf16* brow0 = vp + (long long)((eb0 + 0) * 16 + l15) * NPIX;
  const __bf16* brow1 = vp + (long long)((eb0 + 1) * 16 + l15) * NPIX;
  for (int kk = 0; kk < NPIX; kk += 32) {
    v16bf a  = frag_from(arow + kk + half * 8, arow + kk + 16 + half * 8);
    v16bf b0 = frag_from(brow0 + kk + half * 8, brow0 + kk + 16 + half * 8);
    v16bf b1 = frag_from(brow1 + kk + half * 8, brow1 + kk + 16 + half * 8);
    acc[0] = wmma_bf16(a, b0, acc[0]);
    acc[1] = wmma_bf16(a, b1, acc[1]);
  }
  for (int t = 0; t < 2; ++t) {
    int e = (eb0 + t) * 16 + l15;
    int dbase = dblk * 16 + half * 8;
    for (int j = 0; j < 8; ++j) sCtx[e * 72 + dbase + j] = (__bf16)acc[t][j];
  }
  __syncthreads();

  // ---- phase 2: out2 = context^T @ q   (M=64, K=64, N swept 128-wide)
  __bf16* op = out2 + ((long long)b * CDIM + h * 64) * NPIX;
  const int mblk = wave >> 1;
  const int nb0 = (wave & 1) * 4;

  for (int n0 = 0; n0 < NPIX; n0 += 128) {
    // stage q tile (64 x 128) transposed into sQ[n][d]
    for (int i = 0; i < 4; ++i) {
      int slot = tid + i * 256;        // 0..1023
      int d = slot >> 4;               // 0..63
      int nc = (slot & 15) * 8;        // 0..120
      BF16x8 u; u.v = *(const v8bf*)&qp[(long long)d * NPIX + n0 + nc];
      for (int j = 0; j < 8; ++j) sQ[(nc + j) * 72 + d] = u.s[j];
    }
    __syncthreads();

    v8f a2[4];
    for (int ni = 0; ni < 4; ++ni) a2[ni] = zero8();
    for (int kk2 = 0; kk2 < 2; ++kk2) {
      const __bf16* abase = &sCtx[(mblk * 16 + l15) * 72 + kk2 * 32];
      v16bf ctxf = frag_from(abase + half * 8, abase + 16 + half * 8);
      for (int ni = 0; ni < 4; ++ni) {
        const __bf16* qb = &sQ[((nb0 + ni) * 16 + l15) * 72 + kk2 * 32];
        v16bf bq = frag_from(qb + half * 8, qb + 16 + half * 8);
        a2[ni] = wmma_bf16(ctxf, bq, a2[ni]);
      }
    }
    for (int ni = 0; ni < 4; ++ni) {
      int e = mblk * 16 + half * 8;
      int c = n0 + (nb0 + ni) * 16 + l15;
      for (int j = 0; j < 8; ++j)
        op[(long long)(e + j) * NPIX + c] = (__bf16)a2[ni][j];
    }
    __syncthreads();
  }
}

// ---------------------------------------------------------------------------
extern "C" void kernel_launch(void* const* d_in, const int* in_sizes, int n_in,
                              void* d_out, int out_size, void* d_ws, size_t ws_size,
                              hipStream_t stream) {
  const float* x     = (const float*)d_in[0];  // (8,512,64,64)
  const float* w_qkv = (const float*)d_in[1];  // (1536,512)
  const float* w_out = (const float*)d_in[2];  // (512,512)
  const float* b_out = (const float*)d_in[3];  // (512,)
  float* out = (float*)d_out;                  // (8,512,64,64)

  __bf16* qkv  = (__bf16*)d_ws;                                    // 96 MiB
  __bf16* out2 = (__bf16*)((char*)d_ws +
                 (size_t)8 * QKVC * NPIX * sizeof(__bf16));        // +32 MiB

  // 1) qkv[b] = w_qkv (1536x512) @ x[b] (512x4096)  -> bf16
  dim3 g1(NPIX / 128, QKVC / 128, 8);
  wmma_gemm<false, false><<<g1, 256, 0, stream>>>(
      w_qkv, 0LL, x, (long long)CDIM * NPIX, qkv, (long long)QKVC * NPIX,
      nullptr, CDIM, NPIX);

  // 2) softmax over n on k channels (in place; rewritten by (1) every call)
  softmax_inplace<<<512, 256, 0, stream>>>(qkv);

  // 3) context + out projection-input, one workgroup per (b,h)
  attn_fused<<<64, 256, 0, stream>>>(qkv, out2);

  // 4) out[b] = w_out (512x512) @ out2[b] (512x4096) + b_out  -> f32
  dim3 g4(NPIX / 128, CDIM / 128, 8);
  wmma_gemm<true, true><<<g4, 256, 0, stream>>>(
      w_out, 0LL, out2, (long long)CDIM * NPIX, out, (long long)CDIM * NPIX,
      b_out, CDIM, NPIX);
}
